// GAT_69569880261280
// MI455X (gfx1250) — compile-verified, matrix operand order checked by reading, not run on hardware
//
#include <hip/hip_runtime.h>
#include <hip/hip_bf16.h>

// ---------------- problem constants ----------------
#define NN      100000
#define EE      1600000
#define IN_F    128
#define OUT_F   32
#define HEADS   6
#define HC      (HEADS * OUT_F)      // 192
#define STRIPS  (NN / 16)            // 6250 (N divisible by 16)
#define NEG_SLOPE 0.2f

#define NTILES  (12 * 4)             // 12 column tiles x 4 k-steps
#define WPK_ELEMS (NTILES * 32 * 16) // 24576 bf16 = 48 KB packed W

typedef __attribute__((ext_vector_type(16))) __bf16 v16bf;
typedef __attribute__((ext_vector_type(8)))  float  v8f;

// ---------------- helpers ----------------
__device__ __forceinline__ void atomicMaxFloat(float* addr, float val) {
  // monotone float<->int bit mapping: works with init value -inf
  if (val >= 0.0f) {
    atomicMax((int*)addr, __float_as_int(val));
  } else {
    atomicMin((unsigned int*)addr, (unsigned int)__float_as_int(val));
  }
}

// ---------------- kernels ----------------

// Pack W (f32, 128x192) directly into the WMMA B-operand register layout:
// Wpk[tile][lane][e] where tile = ct*4+kb; lane holds column c0+(lane&15),
// lane half selects K-half; e<8 -> K=khalf+e, e>=8 -> K=khalf+e+8.
__global__ void gat_pack_w(const float* __restrict__ W, __bf16* __restrict__ Wpk) {
  int t = blockIdx.x * blockDim.x + threadIdx.x;
  if (t >= WPK_ELEMS) return;
  int e    = t & 15;
  int lane = (t >> 4) & 31;
  int tile = t >> 9;
  int ct = tile >> 2, kb = tile & 3;
  int ncol  = lane & 15;
  int khalf = (lane & 16) ? 8 : 0;
  int kloc  = (e < 8) ? (khalf + e) : (khalf + e + 8);
  int k = kb * 32 + kloc;
  int c = ct * 16 + ncol;
  Wpk[t] = (__bf16)W[(size_t)k * HC + c];
}

__global__ void gat_fill_neginf(float* __restrict__ p, int n) {
  int t = blockIdx.x * blockDim.x + threadIdx.x;
  if (t < n) p[t] = -__builtin_inff();
}

__device__ __forceinline__ void cvt8(const float4 a, const float4 b, v16bf& v, const int base) {
  v[base + 0] = (__bf16)a.x; v[base + 1] = (__bf16)a.y;
  v[base + 2] = (__bf16)a.z; v[base + 3] = (__bf16)a.w;
  v[base + 4] = (__bf16)b.x; v[base + 5] = (__bf16)b.y;
  v[base + 6] = (__bf16)b.z; v[base + 7] = (__bf16)b.w;
}

// Projection GEMM: xp[N,192] = x[N,128] @ W[128,192] via v_wmma_f32_16x16x32_bf16.
// One wave per 16-row strip; packed W staged in LDS once per workgroup; A tile
// (16x32 bf16/k-step) kept in regs and reused across all 12 column tiles.
__global__ __launch_bounds__(256)
void gat_proj_wmma(const float* __restrict__ x, const __bf16* __restrict__ Wpk,
                   float* __restrict__ xp) {
  __shared__ __align__(16) __bf16 sW[WPK_ELEMS];   // 48 KB

  // cooperative 128-bit copy of packed W into LDS (all 256 threads)
  {
    const uint4* src = (const uint4*)Wpk;
    uint4* dst = (uint4*)sW;
#pragma unroll
    for (int i = 0; i < (WPK_ELEMS * 2 / 16) / 256; ++i)   // 3072 uint4 / 256 = 12
      dst[threadIdx.x + i * 256] = src[threadIdx.x + i * 256];
  }
  __syncthreads();

  const int wave = (int)((blockIdx.x * blockDim.x + threadIdx.x) >> 5);
  const int lane = threadIdx.x & 31;
  if (wave >= STRIPS) return;                  // uniform per wave -> EXEC stays all-1s

  const int r0    = wave * 16;
  const int mrow  = lane & 15;
  const int khalf = (lane < 16) ? 0 : 8;       // ISA 16-bit A layout: lane half selects K-half

  // ---- load A: 4 k-steps of 16x32 bf16 (vector f32 loads, cvt to bf16) ----
  v16bf a[4];
  const float* xrow = x + (size_t)(r0 + mrow) * IN_F;
#pragma unroll
  for (int kb = 0; kb < 4; ++kb) {
    float4 f0 = *(const float4*)(xrow + kb * 32 + khalf);
    float4 f1 = *(const float4*)(xrow + kb * 32 + khalf + 4);
    float4 f2 = *(const float4*)(xrow + kb * 32 + 16 + khalf);
    float4 f3 = *(const float4*)(xrow + kb * 32 + 16 + khalf + 4);
    v16bf av;
    cvt8(f0, f1, av, 0);
    cvt8(f2, f3, av, 8);
    a[kb] = av;
  }

  const int ncol  = lane & 15;
  const int rbase = (lane < 16) ? 0 : 8;       // C layout: lanes 16-31 hold M=8..15

#pragma unroll 1
  for (int ct = 0; ct < 12; ++ct) {
    const int c0 = ct * 16;
    v8f acc = {};
#pragma unroll
    for (int kb = 0; kb < 4; ++kb) {
      // B tile: one 32-byte contiguous LDS read per lane (pre-packed layout)
      const v16bf* bt = (const v16bf*)&sW[((ct * 4 + kb) * 32 + lane) * 16];
      v16bf bv = *bt;
      acc = __builtin_amdgcn_wmma_f32_16x16x32_bf16(
          /*neg_a=*/false, a[kb], /*neg_b=*/false, bv,
          /*c_mod=*/(short)0, acc, /*reuse_a=*/false, /*reuse_b=*/false);
    }
#pragma unroll
    for (int v = 0; v < 8; ++v) {
      xp[(size_t)(r0 + rbase + v) * HC + c0 + ncol] = acc[v];
    }
  }
}

// Per-node attention logit halves: a_s[n,h], a_d[n,h]
__global__ void gat_node_logits(const float* __restrict__ xp,
                                const float* __restrict__ att_src,
                                const float* __restrict__ att_dst,
                                float* __restrict__ a_s, float* __restrict__ a_d) {
  int t = blockIdx.x * blockDim.x + threadIdx.x;
  if (t >= NN * HEADS) return;
  int nod = t / HEADS, h = t % HEADS;
  const float* v = xp + (size_t)nod * HC + h * OUT_F;
  float ss = 0.f, sd = 0.f;
#pragma unroll
  for (int c = 0; c < OUT_F; ++c) {
    float xv = v[c];
    ss += xv * att_src[h * OUT_F + c];
    sd += xv * att_dst[h * OUT_F + c];
  }
  a_s[t] = ss;
  a_d[t] = sd;
}

// Per-edge leaky-relu logits + segment max (atomic) into mbuf[dst,h]
__global__ void gat_edge_logits(const long long* __restrict__ ei,
                                const float* __restrict__ a_s,
                                const float* __restrict__ a_d,
                                float* __restrict__ ebuf, float* __restrict__ mbuf) {
  int e = blockIdx.x * blockDim.x + threadIdx.x;
  if (e >= EE) return;
  int s = (int)ei[e];
  int d = (int)ei[EE + e];
#pragma unroll
  for (int h = 0; h < HEADS; ++h) {
    float t = a_s[s * HEADS + h] + a_d[d * HEADS + h];
    t = (t > 0.f) ? t : NEG_SLOPE * t;
    ebuf[(size_t)e * HEADS + h] = t;
    atomicMaxFloat(&mbuf[d * HEADS + h], t);
  }
}

// exp(e - m[dst]) in place, segment sum (atomic) into sbuf[dst,h]
__global__ void gat_edge_exp(const long long* __restrict__ ei,
                             const float* __restrict__ mbuf,
                             float* __restrict__ ebuf, float* __restrict__ sbuf) {
  int e = blockIdx.x * blockDim.x + threadIdx.x;
  if (e >= EE) return;
  int d = (int)ei[EE + e];
#pragma unroll
  for (int h = 0; h < HEADS; ++h) {
    float p = __expf(ebuf[(size_t)e * HEADS + h] - mbuf[d * HEADS + h]);
    ebuf[(size_t)e * HEADS + h] = p;
    atomicAdd(&sbuf[d * HEADS + h], p);
  }
}

// Weighted scatter-add: one wave per edge, lane = channel within head (coalesced 128B)
__global__ __launch_bounds__(256)
void gat_aggregate(const long long* __restrict__ ei, const float* __restrict__ xp,
                   const float* __restrict__ ebuf, const float* __restrict__ sbuf,
                   float* __restrict__ acc) {
  int wv = (int)((blockIdx.x * blockDim.x + threadIdx.x) >> 5);
  int lane = threadIdx.x & 31;
  if (wv >= EE) return;                       // uniform per wave
  int s = (int)ei[wv];
  int d = (int)ei[EE + wv];
  const float* xs = xp + (size_t)s * HC;
  float* ad = acc + (size_t)d * HC;
#pragma unroll
  for (int h = 0; h < HEADS; ++h) {
    float alpha = ebuf[(size_t)wv * HEADS + h] / (sbuf[d * HEADS + h] + 1e-16f);
    atomicAdd(&ad[h * OUT_F + lane], xs[h * OUT_F + lane] * alpha);
  }
}

// Head mean + bias + SELU -> out[N,32]
__global__ void gat_finalize(const float* __restrict__ acc, const float* __restrict__ bias,
                             float* __restrict__ out) {
  int t = blockIdx.x * blockDim.x + threadIdx.x;
  if (t >= NN * OUT_F) return;
  int c = t & (OUT_F - 1);
  int nod = t >> 5;
  float sum = 0.f;
#pragma unroll
  for (int h = 0; h < HEADS; ++h) sum += acc[(size_t)nod * HC + h * OUT_F + c];
  float v = sum * (1.0f / HEADS) + bias[c];
  const float kScale = 1.0507009873554805f;
  const float kAlpha = 1.6732632423543772f;
  out[t] = (v > 0.f) ? kScale * v : kScale * kAlpha * (__expf(v) - 1.f);
}

// ---------------- launch ----------------
extern "C" void kernel_launch(void* const* d_in, const int* in_sizes, int n_in,
                              void* d_out, int out_size, void* d_ws, size_t ws_size,
                              hipStream_t stream) {
  const float*     x       = (const float*)d_in[0];
  const long long* ei      = (const long long*)d_in[1];   // int64 edge_index [2,E]
  const float*     W       = (const float*)d_in[2];
  const float*     att_src = (const float*)d_in[3];
  const float*     att_dst = (const float*)d_in[4];
  const float*     bias    = (const float*)d_in[5];
  float*           out     = (float*)d_out;

  // workspace carve-up (all offsets 256B aligned), total ~201.7 MB
  char* ws = (char*)d_ws;
  float*  xp   = (float*)(ws + 0);            // N*192 f32 = 76.8 MB
  float*  acc  = (float*)(ws + 76800000);     // N*192 f32 = 76.8 MB
  float*  a_s  = (float*)(ws + 153600000);    // N*6  f32 = 2.4 MB
  float*  a_d  = (float*)(ws + 156000000);    // N*6
  float*  mbuf = (float*)(ws + 158400000);    // N*6
  float*  sbuf = (float*)(ws + 160800000);    // N*6
  float*  ebuf = (float*)(ws + 163200000);    // E*6  f32 = 38.4 MB
  __bf16* Wpk  = (__bf16*)(ws + 201600000);   // 24576 bf16 packed = 48 KB

  // per-call re-init (graph-capture safe)
  hipMemsetAsync(acc,  0, (size_t)NN * HC * sizeof(float), stream);
  hipMemsetAsync(sbuf, 0, (size_t)NN * HEADS * sizeof(float), stream);
  gat_fill_neginf<<<(NN * HEADS + 255) / 256, 256, 0, stream>>>(mbuf, NN * HEADS);

  gat_pack_w<<<(WPK_ELEMS + 255) / 256, 256, 0, stream>>>(W, Wpk);
  gat_proj_wmma<<<(STRIPS * 32 + 255) / 256, 256, 0, stream>>>(x, Wpk, xp);
  gat_node_logits<<<(NN * HEADS + 255) / 256, 256, 0, stream>>>(xp, att_src, att_dst, a_s, a_d);
  gat_edge_logits<<<(EE + 255) / 256, 256, 0, stream>>>(ei, a_s, a_d, ebuf, mbuf);
  gat_edge_exp<<<(EE + 255) / 256, 256, 0, stream>>>(ei, mbuf, ebuf, sbuf);
  gat_aggregate<<<(EE * 32) / 256, 256, 0, stream>>>(ei, xp, ebuf, sbuf, acc);
  gat_finalize<<<(NN * OUT_F + 255) / 256, 256, 0, stream>>>(acc, bias, out);
}
